// ContrastiveWSD_19825569038572
// MI455X (gfx1250) — matrix operand order repaired and checked by reading, not compiled, not run on hardware
//
#include <hip/hip_runtime.h>
#include <hip/hip_bf16.h>
#include <math.h>

#define N_NODES   50000
#define N_EDGES   400000
#define BATCH     512
#define D_ENC     768
#define HID       256
#define LABELS    4096
#define NE_TOT    (N_EDGES + N_NODES)
#define LN_EPS    1e-5f
#define NEG_SLOPE 0.2f

typedef __bf16 bf16;
typedef __attribute__((ext_vector_type(16))) bf16  v16bf;
typedef __attribute__((ext_vector_type(8)))  bf16  v8bf;
typedef __attribute__((ext_vector_type(8)))  float v8f;

// ---------------- helpers ----------------

__device__ __forceinline__ unsigned f2ord(float x) {
    unsigned u = __float_as_uint(x);
    return (u >> 31) ? ~u : (u | 0x80000000u);
}
__device__ __forceinline__ float ord2f(unsigned u) {
    u = (u >> 31) ? (u & 0x7FFFFFFFu) : ~u;
    return __uint_as_float(u);
}
__device__ __forceinline__ float gelu_exact(float y) {
    return 0.5f * y * (1.0f + erff(y * 0.70710678118654752440f));
}
__device__ __forceinline__ float wave_sum(float v) {
    #pragma unroll
    for (int o = 16; o > 0; o >>= 1) v += __shfl_xor(v, o, 32);
    return v;
}

// async 16B copy global -> LDS (CDNA5 GLOBAL_LOAD_ASYNC_TO_LDS_B128, ASYNCcnt-tracked)
__device__ __forceinline__ void async_g2l_b128(unsigned ldsOff, const void* gptr) {
    unsigned long long ga = (unsigned long long)(uintptr_t)gptr;
    asm volatile("global_load_async_to_lds_b128 %0, %1, off"
                 :: "v"(ldsOff), "v"(ga) : "memory");
}
__device__ __forceinline__ void wait_async0() {
    asm volatile("s_wait_asynccnt 0" ::: "memory");
}

// ---------------- precision conversion / weight packing ----------------

__global__ void k_f32_to_bf16(const float* __restrict__ x, bf16* __restrict__ y, int n) {
    int i = blockIdx.x * blockDim.x + threadIdx.x;
    if (i < n) y[i] = (bf16)x[i];
}

// Pack W[K x 256] (row-major f32) into B-fragment-major bf16:
// P[((nt*KS + ks)*32 + lane)*16 + e] = W[(ks*32 + (e<8 ? 8*(lane>>4) + e : 16 + 8*(lane>>4) + e-8))*256 + nt*16 + (lane&15)]
__global__ void k_pack_w(const float* __restrict__ W, bf16* __restrict__ P, int K) {
    int idx = blockIdx.x * blockDim.x + threadIdx.x;
    int ksteps = K >> 5;
    int total = 16 * ksteps * 32 * 16;
    if (idx >= total) return;
    int e    = idx & 15;
    int lane = (idx >> 4) & 31;
    int rest = idx >> 9;               // nt*ksteps + ks
    int nt   = rest / ksteps;
    int ks   = rest - nt * ksteps;
    int k = ks * 32 + ((e < 8) ? (((lane >> 4) << 3) + e) : (16 + ((lane >> 4) << 3) + (e - 8)));
    int n = (nt << 4) + (lane & 15);
    P[idx] = (bf16)W[(size_t)k * HID + n];
}

// ---------------- WMMA GEMM: C[M x 256] = A[M x K](bf16) @ Wpack + bias ----------------
// Block: 256 threads = 8 waves; block computes 32 rows x 256 cols (2 M-tiles).
// Wave w computes N-tiles {2w,2w+1} x M-tiles {0,1}: 4 WMMAs per K-step, B reused across M.

template <int K, bool FUSE_LN>
__global__ __launch_bounds__(256)
void k_gemm(const bf16* __restrict__ A,      // M x K row-major bf16
            const bf16* __restrict__ Bp,     // packed weights
            const float* __restrict__ bias,  // [256]
            const float* __restrict__ gamma, // [256] (LN) or nullptr
            const float* __restrict__ beta,  // [256] (LN) or nullptr
            float* __restrict__ outF,        // M x 256 f32
            bf16*  __restrict__ outB,        // M x 256 bf16 or nullptr
            int M)
{
    constexpr int KS  = K / 32;
    constexpr int LDA = K + 8;                       // bf16 elems per LDS row (pad)
    constexpr unsigned ABYTES = 32u * LDA * 2u;      // A tile bytes
    constexpr unsigned CBYTES = 32u * 264u * 4u;     // LN epilogue tile bytes
    constexpr unsigned SBYTES = FUSE_LN ? (ABYTES > CBYTES ? ABYTES : CBYTES) : ABYTES;
    __shared__ __align__(16) char smem[SBYTES];
    bf16*  sA = (bf16*)smem;
    float* sC = (float*)smem;                        // reused after barrier (LN only)

    const int tid   = threadIdx.x;
    const int lane  = tid & 31;
    const int wave  = tid >> 5;
    const int mBase = blockIdx.x * 32;

    // async stage of the 32 x K A-tile into LDS (16B async copies, ASYNCcnt)
    {
        const int cpr = K / 8;                       // 8-bf16 chunks per row
        for (int c = tid; c < 32 * cpr; c += 256) {
            int r  = c / cpr;
            int kk = (c - r * cpr) * 8;
            int sr = mBase + r; if (sr >= M) sr = M - 1;   // clamp tail rows
            unsigned ldsOff = (unsigned)(uintptr_t)&sA[r * LDA + kk];
            async_g2l_b128(ldsOff, &A[(size_t)sr * K + kk]);
        }
    }
    wait_async0();
    __syncthreads();

    const int row = lane & 15;
    const int hi  = lane >> 4;
    const int nt0 = wave * 2, nt1 = wave * 2 + 1;
    const bf16* bp0 = Bp + (size_t)nt0 * KS * 512;
    const bf16* bp1 = Bp + (size_t)nt1 * KS * 512;

    v8f acc00 = {}, acc01 = {}, acc10 = {}, acc11 = {};   // [mtile][ntile]

    #pragma unroll 2
    for (int ks = 0; ks < KS; ++ks) {
        // A fragments: two contiguous 8-elem runs per lane (ISA 16-bit A 16x32 layout)
        v8bf a0 = *(const v8bf*)&sA[row * LDA + ks * 32 + 8 * hi];
        v8bf a1 = *(const v8bf*)&sA[row * LDA + ks * 32 + 16 + 8 * hi];
        v8bf c0 = *(const v8bf*)&sA[(16 + row) * LDA + ks * 32 + 8 * hi];
        v8bf c1 = *(const v8bf*)&sA[(16 + row) * LDA + ks * 32 + 16 + 8 * hi];
        v16bf af0, af1;
        #pragma unroll
        for (int i = 0; i < 8; ++i) {
            af0[i] = a0[i]; af0[8 + i] = a1[i];
            af1[i] = c0[i]; af1[8 + i] = c1[i];
        }
        v16bf b0 = *(const v16bf*)&bp0[((size_t)ks * 32 + lane) * 16];
        v16bf b1 = *(const v16bf*)&bp1[((size_t)ks * 32 + lane) * 16];

        acc00 = __builtin_amdgcn_wmma_f32_16x16x32_bf16(false, af0, false, b0, (short)0, acc00, false, false);
        acc10 = __builtin_amdgcn_wmma_f32_16x16x32_bf16(false, af1, false, b0, (short)0, acc10, false, false);
        acc01 = __builtin_amdgcn_wmma_f32_16x16x32_bf16(false, af0, false, b1, (short)0, acc01, false, false);
        acc11 = __builtin_amdgcn_wmma_f32_16x16x32_bf16(false, af1, false, b1, (short)0, acc11, false, false);
    }

    const int col0 = nt0 * 16 + (lane & 15);
    const int col1 = nt1 * 16 + (lane & 15);

    if constexpr (!FUSE_LN) {
        #pragma unroll
        for (int v = 0; v < 8; ++v) {
            int m0 = mBase + v + 8 * hi;
            int m1 = m0 + 16;
            if (m0 < M) {
                outF[(size_t)m0 * HID + col0] = acc00[v] + bias[col0];
                outF[(size_t)m0 * HID + col1] = acc01[v] + bias[col1];
            }
            if (m1 < M) {
                outF[(size_t)m1 * HID + col0] = acc10[v] + bias[col0];
                outF[(size_t)m1 * HID + col1] = acc11[v] + bias[col1];
            }
        }
    } else {
        __syncthreads();                              // done with sA; reuse as sC
        #pragma unroll
        for (int v = 0; v < 8; ++v) {
            int r = v + 8 * hi;
            sC[r * 264 + col0]        = acc00[v] + bias[col0];
            sC[r * 264 + col1]        = acc01[v] + bias[col1];
            sC[(r + 16) * 264 + col0] = acc10[v] + bias[col0];
            sC[(r + 16) * 264 + col1] = acc11[v] + bias[col1];
        }
        __syncthreads();
        // wave handles 4 rows; lane handles 8 cols (c = lane + 32j) -> LN + exact GELU
        #pragma unroll
        for (int rr = 0; rr < 4; ++rr) {
            int r = wave * 4 + rr;
            int m = mBase + r;
            if (m >= M) continue;
            float vals[8], s = 0.f;
            #pragma unroll
            for (int j = 0; j < 8; ++j) { vals[j] = sC[r * 264 + lane + 32 * j]; s += vals[j]; }
            float mu = wave_sum(s) * (1.0f / 256.0f);
            float vv = 0.f;
            #pragma unroll
            for (int j = 0; j < 8; ++j) { float d = vals[j] - mu; vv += d * d; }
            float rstd = rsqrtf(wave_sum(vv) * (1.0f / 256.0f) + LN_EPS);
            #pragma unroll
            for (int j = 0; j < 8; ++j) {
                int c = lane + 32 * j;
                float y  = (vals[j] - mu) * rstd * gamma[c] + beta[c];
                float ge = gelu_exact(y);
                outF[(size_t)m * HID + c] = ge;
                if (outB) outB[(size_t)m * HID + c] = (bf16)ge;
            }
        }
    }
}

// ---------------- GATv2 graph phase ----------------

__global__ void k_gat_init(unsigned* __restrict__ mEnc, float* __restrict__ denom,
                           float* __restrict__ out, const float* __restrict__ bias) {
    size_t i = (size_t)blockIdx.x * blockDim.x + threadIdx.x;
    if (i < (size_t)N_NODES * HID) out[i] = bias[i & (HID - 1)];
    if (i < N_NODES) { mEnc[i] = f2ord(-INFINITY); denom[i] = 0.f; }
}

// wave per edge: e = sum_c att[c]*leaky_relu(xl[src][c]+xr[dst][c]); segment-max via ordered atomicMax
__global__ __launch_bounds__(256)
void k_edge_logits(const int* __restrict__ esrc, const int* __restrict__ edst,
                   const float* __restrict__ xl, const float* __restrict__ xr,
                   const float* __restrict__ att,
                   float* __restrict__ eAll, unsigned* __restrict__ mEnc)
{
    int w    = (blockIdx.x * blockDim.x + threadIdx.x) >> 5;
    int lane = threadIdx.x & 31;
    if (w >= NE_TOT) return;
    int s, d;
    if (w < N_EDGES) { s = esrc[w]; d = edst[w]; } else { s = w - N_EDGES; d = s; }
    const float* pl = xl + (size_t)s * HID;
    const float* pr = xr + (size_t)d * HID;
    float acc = 0.f;
    #pragma unroll
    for (int j = 0; j < 8; ++j) {
        int c = lane + 32 * j;
        float v = pl[c] + pr[c];
        v = (v > 0.f) ? v : v * NEG_SLOPE;
        acc += att[c] * v;
    }
    acc = wave_sum(acc);
    if (lane == 0) { eAll[w] = acc; atomicMax(&mEnc[d], f2ord(acc)); }
}

__global__ void k_edge_exp(const int* __restrict__ edst, const float* __restrict__ eAll,
                           const unsigned* __restrict__ mEnc,
                           float* __restrict__ ee, float* __restrict__ denom)
{
    int i = blockIdx.x * blockDim.x + threadIdx.x;
    if (i >= NE_TOT) return;
    int d = (i < N_EDGES) ? edst[i] : (i - N_EDGES);
    float v = expf(eAll[i] - ord2f(mEnc[d]));
    ee[i] = v;
    atomicAdd(&denom[d], v);
}

// wave per edge: out[dst] += alpha * xl[src]
__global__ __launch_bounds__(256)
void k_edge_aggr(const int* __restrict__ esrc, const int* __restrict__ edst,
                 const float* __restrict__ xl, const float* __restrict__ ee,
                 const float* __restrict__ denom, float* __restrict__ out)
{
    int w    = (blockIdx.x * blockDim.x + threadIdx.x) >> 5;
    int lane = threadIdx.x & 31;
    if (w >= NE_TOT) return;
    int s, d;
    if (w < N_EDGES) { s = esrc[w]; d = edst[w]; } else { s = w - N_EDGES; d = s; }
    float alpha = ee[w] / denom[d];
    const float* pl = xl + (size_t)s * HID;
    float*       po = out + (size_t)d * HID;
    #pragma unroll
    for (int j = 0; j < 8; ++j) {
        int c = lane + 32 * j;
        atomicAdd(&po[c], alpha * pl[c]);
    }
}

// wave per row LN + GELU; optional bf16 mirror and optional slice copy into d_out
__global__ __launch_bounds__(256)
void k_ln_gelu(const float* __restrict__ x,
               const float* __restrict__ gamma, const float* __restrict__ beta,
               float* __restrict__ outF, bf16* __restrict__ outB,
               float* __restrict__ out2, int rows2, int nrows)
{
    int r    = (blockIdx.x * blockDim.x + threadIdx.x) >> 5;
    int lane = threadIdx.x & 31;
    if (r >= nrows) return;
    const float* px = x + (size_t)r * HID;
    float vals[8], s = 0.f;
    #pragma unroll
    for (int j = 0; j < 8; ++j) { vals[j] = px[lane + 32 * j]; s += vals[j]; }
    float mu = wave_sum(s) * (1.0f / 256.0f);
    float vv = 0.f;
    #pragma unroll
    for (int j = 0; j < 8; ++j) { float d = vals[j] - mu; vv += d * d; }
    float rstd = rsqrtf(wave_sum(vv) * (1.0f / 256.0f) + LN_EPS);
    #pragma unroll
    for (int j = 0; j < 8; ++j) {
        int c = lane + 32 * j;
        float y  = (vals[j] - mu) * rstd * gamma[c] + beta[c];
        float ge = gelu_exact(y);
        if (outF) outF[(size_t)r * HID + c] = ge;
        if (outB) outB[(size_t)r * HID + c] = (bf16)ge;
        if (out2 && r < rows2) out2[(size_t)r * HID + c] = ge;
    }
}

// ---------------- host launch ----------------

extern "C" void kernel_launch(void* const* d_in, const int* in_sizes, int n_in,
                              void* d_out, int out_size, void* d_ws, size_t ws_size,
                              hipStream_t stream)
{
    const float* word_feats  = (const float*)d_in[0];
    const float* gloss_feats = (const float*)d_in[1];
    const int*   edges       = (const int*)d_in[2];
    const float* Ww = (const float*)d_in[4];
    const float* bw = (const float*)d_in[5];
    const float* gw1 = (const float*)d_in[6];
    const float* bw1 = (const float*)d_in[7];
    const float* Wc = (const float*)d_in[8];
    const float* bc = (const float*)d_in[9];
    const float* gc1 = (const float*)d_in[10];
    const float* bc1 = (const float*)d_in[11];
    const float* g2_Wl = (const float*)d_in[12];
    const float* g2_Wr = (const float*)d_in[13];
    const float* g2_bl = (const float*)d_in[14];
    const float* g2_br = (const float*)d_in[15];
    const float* g2_att = (const float*)d_in[16];
    const float* g2_bias = (const float*)d_in[17];
    const float* gc2 = (const float*)d_in[18];
    const float* bc2 = (const float*)d_in[19];
    const float* g3_Wl = (const float*)d_in[20];
    const float* g3_Wr = (const float*)d_in[21];
    const float* g3_bl = (const float*)d_in[22];
    const float* g3_br = (const float*)d_in[23];
    const float* g3_att = (const float*)d_in[24];
    const float* g3_bias = (const float*)d_in[25];
    const float* gc3 = (const float*)d_in[26];
    const float* bc3 = (const float*)d_in[27];

    const int* esrc = edges;
    const int* edst = edges + N_EDGES;

    // workspace arena
    char* ws = (char*)d_ws;
    auto alloc = [&](size_t bytes) -> void* {
        void* p = ws;
        ws += (bytes + 255) & ~(size_t)255;
        return p;
    };
    bf16* glossB = (bf16*)alloc((size_t)N_NODES * D_ENC * 2);
    bf16* wordB  = (bf16*)alloc((size_t)BATCH * D_ENC * 2);
    bf16* packWw = (bf16*)alloc((size_t)16 * 24 * 512 * 2);
    bf16* packWc = (bf16*)alloc((size_t)16 * 24 * 512 * 2);
    bf16* pack2l = (bf16*)alloc((size_t)16 * 8 * 512 * 2);
    bf16* pack2r = (bf16*)alloc((size_t)16 * 8 * 512 * 2);
    bf16* pack3l = (bf16*)alloc((size_t)16 * 8 * 512 * 2);
    bf16* pack3r = (bf16*)alloc((size_t)16 * 8 * 512 * 2);
    float* xl    = (float*)alloc((size_t)N_NODES * HID * 4);
    float* xr    = (float*)alloc((size_t)N_NODES * HID * 4);
    float* accum = (float*)alloc((size_t)N_NODES * HID * 4);
    float* gF    = (float*)alloc((size_t)N_NODES * HID * 4);
    bf16*  gB    = (bf16*)alloc((size_t)N_NODES * HID * 2);
    float* eAll  = (float*)alloc((size_t)NE_TOT * 4);
    float* ee    = (float*)alloc((size_t)NE_TOT * 4);
    unsigned* mEnc = (unsigned*)alloc((size_t)N_NODES * 4);
    float* denom = (float*)alloc((size_t)N_NODES * 4);

    auto cdiv = [](long long a, long long b) { return (int)((a + b - 1) / b); };

    // 1) convert inputs to bf16
    k_f32_to_bf16<<<cdiv((long long)N_NODES * D_ENC, 256), 256, 0, stream>>>(gloss_feats, glossB, N_NODES * D_ENC);
    k_f32_to_bf16<<<cdiv((long long)BATCH * D_ENC, 256), 256, 0, stream>>>(word_feats, wordB, BATCH * D_ENC);

    // 2) pack weights
    k_pack_w<<<cdiv(D_ENC * HID, 256), 256, 0, stream>>>(Ww, packWw, D_ENC);
    k_pack_w<<<cdiv(D_ENC * HID, 256), 256, 0, stream>>>(Wc, packWc, D_ENC);
    k_pack_w<<<cdiv(HID * HID, 256), 256, 0, stream>>>(g2_Wl, pack2l, HID);
    k_pack_w<<<cdiv(HID * HID, 256), 256, 0, stream>>>(g2_Wr, pack2r, HID);
    k_pack_w<<<cdiv(HID * HID, 256), 256, 0, stream>>>(g3_Wl, pack3l, HID);
    k_pack_w<<<cdiv(HID * HID, 256), 256, 0, stream>>>(g3_Wr, pack3r, HID);

    float* refined = (float*)d_out;                       // [512 x 256]
    float* gnn_out = (float*)d_out + (size_t)BATCH * HID; // [4096 x 256]

    const int gridWord  = cdiv(BATCH, 32);
    const int gridNodes = cdiv(N_NODES, 32);

    // 3) refined = norm_gelu(word @ Ww + bw)  -> d_out directly
    k_gemm<D_ENC, true><<<gridWord, 256, 0, stream>>>(wordB, packWw, bw, gw1, bw1, refined, (bf16*)nullptr, BATCH);

    // 4) g = norm_gelu(gloss @ Wc + bc)
    k_gemm<D_ENC, true><<<gridNodes, 256, 0, stream>>>(glossB, packWc, bc, gc1, bc1, gF, gB, N_NODES);

    const int edgeWaveBlocks = cdiv((long long)NE_TOT * 32, 256);
    const int edgeThrBlocks  = cdiv(NE_TOT, 256);
    const int initBlocks     = cdiv((long long)N_NODES * HID, 256);
    const int rowWaveBlocks  = cdiv((long long)N_NODES * 32, 256);

    // 5) GAT layer 2
    k_gemm<HID, false><<<gridNodes, 256, 0, stream>>>(gB, pack2l, g2_bl, nullptr, nullptr, xl, (bf16*)nullptr, N_NODES);
    k_gemm<HID, false><<<gridNodes, 256, 0, stream>>>(gB, pack2r, g2_br, nullptr, nullptr, xr, (bf16*)nullptr, N_NODES);
    k_gat_init<<<initBlocks, 256, 0, stream>>>(mEnc, denom, accum, g2_bias);
    k_edge_logits<<<edgeWaveBlocks, 256, 0, stream>>>(esrc, edst, xl, xr, g2_att, eAll, mEnc);
    k_edge_exp<<<edgeThrBlocks, 256, 0, stream>>>(edst, eAll, mEnc, ee, denom);
    k_edge_aggr<<<edgeWaveBlocks, 256, 0, stream>>>(esrc, edst, xl, ee, denom, accum);
    k_ln_gelu<<<rowWaveBlocks, 256, 0, stream>>>(accum, gc2, bc2, gF, gB, (float*)nullptr, 0, N_NODES);

    // 6) GAT layer 3
    k_gemm<HID, false><<<gridNodes, 256, 0, stream>>>(gB, pack3l, g3_bl, nullptr, nullptr, xl, (bf16*)nullptr, N_NODES);
    k_gemm<HID, false><<<gridNodes, 256, 0, stream>>>(gB, pack3r, g3_br, nullptr, nullptr, xr, (bf16*)nullptr, N_NODES);
    k_gat_init<<<initBlocks, 256, 0, stream>>>(mEnc, denom, accum, g3_bias);
    k_edge_logits<<<edgeWaveBlocks, 256, 0, stream>>>(esrc, edst, xl, xr, g3_att, eAll, mEnc);
    k_edge_exp<<<edgeThrBlocks, 256, 0, stream>>>(edst, eAll, mEnc, ee, denom);
    k_edge_aggr<<<edgeWaveBlocks, 256, 0, stream>>>(esrc, edst, xl, ee, denom, accum);
    // final LN+GELU: rows [0, LABELS) also streamed straight into d_out
    k_ln_gelu<<<rowWaveBlocks, 256, 0, stream>>>(accum, gc3, bc3, gF, (bf16*)nullptr, gnn_out, LABELS, N_NODES);

    (void)in_sizes; (void)n_in; (void)out_size; (void)ws_size;
}
